// CostVolume2D_45767171506370
// MI455X (gfx1250) — matrix operand order repaired and checked by reading, not compiled, not run on hardware
//
#include <hip/hip_runtime.h>
#include <hip/hip_bf16.h>

typedef __attribute__((ext_vector_type(16))) __bf16 v16bf;
typedef __attribute__((ext_vector_type(8)))  float  v8f;

// Problem constants (fixed by the reference)
static constexpr int DD = 4;     // search range
static constexpr int NOFF = 81;  // (2*DD+1)^2
static constexpr int CC = 128;
static constexpr int HH = 96;
static constexpr int WW = 224;
static constexpr int NN = 8;
static constexpr int TPX = 16;   // f1 pixels per tile
static constexpr int QPX = 32;   // f2 columns staged in LDS (24 used, padded to 32)

union Frag {
    v16bf v;
    uint4 q[2];
};

__device__ __forceinline__ unsigned short f32_to_bf16_rne(float f) {
    unsigned int u = __float_as_uint(f);
    unsigned int r = (u + 0x7FFFu + ((u >> 16) & 1u)) >> 16;
    return (unsigned short)r;
}

__device__ __forceinline__ unsigned int pack2_bf16(float lo, float hi) {
    return (unsigned int)f32_to_bf16_rne(lo) |
           ((unsigned int)f32_to_bf16_rne(hi) << 16);
}

__global__ __launch_bounds__(32)
void cost_volume2d_wmma_kernel(const float* __restrict__ f1,
                               const float* __restrict__ f2,
                               float* __restrict__ out) {
    __shared__ unsigned short f1s[TPX * CC];  // [p][c], c contiguous
    __shared__ unsigned short f2s[QPX * CC];  // [q][c], c contiguous
    __shared__ float gs[TPX * QPX];           // Gram tile [p][q]

    const int lane = threadIdx.x;          // 0..31 (wave32)
    const int x0 = blockIdx.x * TPX;       // 0..208
    const int y  = blockIdx.y;             // 0..95
    const int n  = blockIdx.z;             // 0..7

    const int HW = HH * WW;

    // ---- Stage f1 tile: 16 px x 128 ch, f32 -> bf16, transpose into [p][c].
    // Each slot = (channel quad, 4-pixel chunk): 4 global b128 loads sharing
    // one address (imm offsets c*HW*4), 4 packed ds_store_b64.
    const float* f1base = f1 + (((size_t)n * CC) * HH + y) * WW + x0;
    for (int i = lane; i < (CC / 4) * 4; i += 32) {   // 128 slots, 4 iters
        const int c  = (i >> 2) * 4;
        const int p0 = (i & 3) * 4;
        const float* src = f1base + (size_t)c * HW + p0;
        const float4 v0 = *reinterpret_cast<const float4*>(src);
        const float4 v1 = *reinterpret_cast<const float4*>(src + HW);
        const float4 v2 = *reinterpret_cast<const float4*>(src + 2 * HW);
        const float4 v3 = *reinterpret_cast<const float4*>(src + 3 * HW);
        uint2* dst0 = reinterpret_cast<uint2*>(&f1s[(p0 + 0) * CC + c]);
        uint2* dst1 = reinterpret_cast<uint2*>(&f1s[(p0 + 1) * CC + c]);
        uint2* dst2 = reinterpret_cast<uint2*>(&f1s[(p0 + 2) * CC + c]);
        uint2* dst3 = reinterpret_cast<uint2*>(&f1s[(p0 + 3) * CC + c]);
        *dst0 = make_uint2(pack2_bf16(v0.x, v1.x), pack2_bf16(v2.x, v3.x));
        *dst1 = make_uint2(pack2_bf16(v0.y, v1.y), pack2_bf16(v2.y, v3.y));
        *dst2 = make_uint2(pack2_bf16(v0.z, v1.z), pack2_bf16(v2.z, v3.z));
        *dst3 = make_uint2(pack2_bf16(v0.w, v1.w), pack2_bf16(v2.w, v3.w));
    }
    __syncthreads();

    // ---- A fragments (f1), kept in registers across all dy / q-tiles.
    // 16-bit A 16x32 layout: lane L holds row m=L%16,
    //   K = 8*(L/16)+0..7 (VGPR0-3) and K = 16+8*(L/16)+0..7 (VGPR4-7).
    Frag afrag[4];
    {
        const int m     = lane & 15;
        const int khalf = (lane >> 4) * 8;   // 0 or 8
#pragma unroll
        for (int cb = 0; cb < 4; ++cb) {
            const unsigned short* src = &f1s[m * CC + cb * 32];
            afrag[cb].q[0] = *reinterpret_cast<const uint4*>(src + khalf);
            afrag[cb].q[1] = *reinterpret_cast<const uint4*>(src + 16 + khalf);
        }
    }

    const float* f2base = f2 + (size_t)(n * CC) * HW;
    const float inv_c = 1.0f / (float)CC;

    for (int dy = 0; dy < 2 * DD + 1; ++dy) {
        const int ry = y + dy - DD;
        const bool rowok = (ry >= 0) && (ry < HH);

        __syncthreads();  // previous iteration fully done with f2s/gs

        // ---- Stage f2 row segment: q = 0..23 maps to x = x0-4 .. x0+19.
        // 8 chunks of 4 px per channel quad (chunks 6,7 forced zero -> pad).
        // Chunks are 16B aligned and never straddle the W boundary.
        for (int i = lane; i < (CC / 4) * 8; i += 32) {  // 256 slots, 8 iters
            const int c  = (i >> 3) * 4;
            const int j  = i & 7;
            const int xb = x0 - DD + 4 * j;
            float4 v0 = make_float4(0.f, 0.f, 0.f, 0.f);
            float4 v1 = v0, v2 = v0, v3 = v0;
            if (rowok && j < 6 && xb >= 0 && (xb + 3) < WW) {
                const float* src = f2base + ((size_t)c * HH + ry) * WW + xb;
                v0 = *reinterpret_cast<const float4*>(src);
                v1 = *reinterpret_cast<const float4*>(src + HW);
                v2 = *reinterpret_cast<const float4*>(src + 2 * HW);
                v3 = *reinterpret_cast<const float4*>(src + 3 * HW);
            }
            const int q = 4 * j;
            uint2* dst0 = reinterpret_cast<uint2*>(&f2s[(q + 0) * CC + c]);
            uint2* dst1 = reinterpret_cast<uint2*>(&f2s[(q + 1) * CC + c]);
            uint2* dst2 = reinterpret_cast<uint2*>(&f2s[(q + 2) * CC + c]);
            uint2* dst3 = reinterpret_cast<uint2*>(&f2s[(q + 3) * CC + c]);
            *dst0 = make_uint2(pack2_bf16(v0.x, v1.x), pack2_bf16(v2.x, v3.x));
            *dst1 = make_uint2(pack2_bf16(v0.y, v1.y), pack2_bf16(v2.y, v3.y));
            *dst2 = make_uint2(pack2_bf16(v0.z, v1.z), pack2_bf16(v2.z, v3.z));
            *dst3 = make_uint2(pack2_bf16(v0.w, v1.w), pack2_bf16(v2.w, v3.w));
        }
        __syncthreads();

        // ---- Gram tiles: G[p, q] = sum_c f1bf[c,p] * f2bf[c,q]
        // 16-bit B 32x16 layout: lane L holds column n=L%16,
        //   K = 8*(L/16)+0..7 (VGPR0-3) and K = 16+8*(L/16)+0..7 (VGPR4-7).
#pragma unroll
        for (int qt = 0; qt < 2; ++qt) {
            v8f acc = {};
            const int qcol = qt * 16 + (lane & 15);
            const int kh   = (lane >> 4) * 8;
#pragma unroll
            for (int cb = 0; cb < 4; ++cb) {
                Frag b;
                const unsigned short* src = &f2s[qcol * CC + cb * 32 + kh];
                b.q[0] = *reinterpret_cast<const uint4*>(src);
                b.q[1] = *reinterpret_cast<const uint4*>(src + 16);
                acc = __builtin_amdgcn_wmma_f32_16x16x32_bf16(
                    false, afrag[cb].v, false, b.v,
                    (short)0, acc, false, false);
            }
            // f32 C/D 16x16 layout: lane L holds col n=L%16,
            // rows m = 8*(L/16) + r for VGPR r.
            const int ncol  = lane & 15;
            const int mbase = (lane >> 4) * 8;
#pragma unroll
            for (int r = 0; r < 8; ++r) {
                gs[(mbase + r) * QPX + qt * 16 + ncol] = acc[r];
            }
        }
        __syncthreads();

        // ---- Extract band: out[dx, p] = G[p, p + dx] / C,  dx = 0..8
        // lanes 0-15 take even dx, lanes 16-31 take odd dx, 5 rounds.
        const int p   = lane & 15;
        const int sel = lane >> 4;
#pragma unroll
        for (int t = 0; t < 5; ++t) {
            int dx = 2 * t + sel;
            if (dx < 2 * DD + 1) {
                float val = gs[p * QPX + p + dx] * inv_c;
                size_t o = ((((size_t)n * NOFF) + (dy * (2 * DD + 1) + dx)) * HH + y)
                               * WW + x0 + p;
                out[o] = val;
            }
        }
    }
}

extern "C" void kernel_launch(void* const* d_in, const int* in_sizes, int n_in,
                              void* d_out, int out_size, void* d_ws, size_t ws_size,
                              hipStream_t stream) {
    const float* f1 = (const float*)d_in[0];
    const float* f2 = (const float*)d_in[1];
    float* out = (float*)d_out;
    dim3 grid(WW / TPX, HH, NN);  // 14 x 96 x 8 workgroups, one wave32 each
    cost_volume2d_wmma_kernel<<<grid, 32, 0, stream>>>(f1, f2, out);
}
